// Layer_61598420959596
// MI455X (gfx1250) — matrix operand order repaired and checked by reading, not compiled, not run on hardware
//
#include <hip/hip_runtime.h>
#include <stdint.h>

// C[16,12288] = A[16,4096] (f32) @ dequant_fp4(codes[4096,12288] int32, s[32,12288] f32)
// Memory-bound: 201MB of codes @ 23.3 TB/s ~= 8.6us floor. Strategy: stream codes
// once (NT loads), v_perm-based byte-LUT dequant (~3 VALU/code), f16 WMMA
// (v_wmma_f32_16x16x32_f16), deterministic split-K LDS reduction.

typedef _Float16 h2   __attribute__((ext_vector_type(2)));
typedef _Float16 v16h __attribute__((ext_vector_type(16)));
typedef float    v8f  __attribute__((ext_vector_type(8)));

#define PERM(hi_, lo_, sel_) __builtin_amdgcn_perm((uint32_t)(hi_), (uint32_t)(lo_), (uint32_t)(sel_))

// v_cvt_pk_rtz_f16_f32 returns __fp16x2; bit-cast to our _Float16x2 type.
__device__ __forceinline__ h2 pkrtz(float a, float b) {
  return __builtin_bit_cast(h2, __builtin_amdgcn_cvt_pkrtz(a, b));
}

// Dequant 4 FP4 codes (each in low byte of an int32) -> two packed-half2 values
// scaled by hs (= 0.5*group_scale broadcast). Sign handled by OR-ing FP4 bit3
// into f16 bit15 (byte-aligned <<4 on the upper-byte lane).
__device__ __forceinline__ void dequad(uint32_t c0, uint32_t c1, uint32_t c2, uint32_t c3,
                                       h2 hs, h2& out01, h2& out23) {
  // pack low bytes: [c3,c2,c1,c0]
  uint32_t p01    = PERM(c1, c0, 0x0C0C0400u);   // [0,0,c1,c0]
  uint32_t p23    = PERM(c3, c2, 0x04000C0Cu);   // [c3,c2,0,0]
  uint32_t packed = p01 | p23;
  uint32_t mg     = packed & 0x07070707u;        // 3-bit magnitude index per byte
  // f16 upper-byte LUT for mags {0,.5,1,1.5, 2,3,4,6} = {00,38,3C,3E, 40,42,44,46}
  uint32_t hib    = PERM(0x46444240u, 0x3E3C3800u, mg);
  hib            |= (packed & 0x08080808u) << 4; // FP4 sign bit -> f16 sign bit
  uint32_t u01    = PERM(hib, hib, 0x010C000Cu); // halves: [f16(c1), f16(c0)]
  uint32_t u23    = PERM(hib, hib, 0x030C020Cu); // halves: [f16(c3), f16(c2)]
  out01 = __builtin_bit_cast(h2, u01) * hs;      // v_pk_mul_f16
  out23 = __builtin_bit_cast(h2, u23) * hs;
}

__global__ __launch_bounds__(256) void marlin_fp4_wmma(
    const float* __restrict__ A, const int* __restrict__ codes,
    const float* __restrict__ s, float* __restrict__ out) {
  constexpr int Kdim = 4096, Ndim = 12288, GS = 128;
  const int lane = threadIdx.x & 31;
  const int wid  = threadIdx.x >> 5;   // 8 waves split K
  const int n0   = blockIdx.x * 32;    // 32-column tile (two 16-wide WMMA N-tiles)
  const int lo16 = lane & 15;
  const int hx   = lane >> 4;

  __shared__ float red[8][512];

  v8f acc0, acc1;
#pragma unroll
  for (int i = 0; i < 8; ++i) { acc0[i] = 0.0f; acc1[i] = 0.0f; }

  const int kw = wid * 512;            // this wave's K chunk: [kw, kw+512)
  for (int g = 0; g < 4; ++g) {        // 4 scale-groups of 128 per chunk
    const int kg  = kw + g * GS;
    const int grp = kg / GS;
    const float sc0 = 0.5f * s[grp * Ndim + n0 + lo16];
    const float sc1 = 0.5f * s[grp * Ndim + n0 + 16 + lo16];
    const _Float16 h0 = (_Float16)sc0, h1 = (_Float16)sc1;
    const h2 hs0 = {h0, h0};
    const h2 hs1 = {h1, h1};

#pragma unroll
    for (int ks = 0; ks < 4; ++ks) {   // K=32 per WMMA step
      const int k = kg + ks * 32;

      // ---- A fragment: 16x32 f16. lanes0-15: K 0-15; lanes16-31: K 16-31
      union { v16h v; h2 h[8]; } af;
      const float* ap = A + lo16 * Kdim + k + 16 * hx;
      const float4 f0 = *(const float4*)(ap +  0);
      const float4 f1 = *(const float4*)(ap +  4);
      const float4 f2 = *(const float4*)(ap +  8);
      const float4 f3 = *(const float4*)(ap + 12);
      af.h[0] = pkrtz(f0.x, f0.y);
      af.h[1] = pkrtz(f0.z, f0.w);
      af.h[2] = pkrtz(f1.x, f1.y);
      af.h[3] = pkrtz(f1.z, f1.w);
      af.h[4] = pkrtz(f2.x, f2.y);
      af.h[5] = pkrtz(f2.z, f2.w);
      af.h[6] = pkrtz(f3.x, f3.y);
      af.h[7] = pkrtz(f3.z, f3.w);

      // ---- B fragments: 32x16 f16 each. lane = column, VGPR j = K pair {2j,2j+1},
      //      lanes16-31 hold K 16-31.  (fragment layout == load pattern; no shuffles)
      const int kb = k + 16 * hx;
      const int* cp0 = codes + (size_t)kb * Ndim + n0 + lo16; // subtile 0
      union { v16h v; h2 h[8]; } b0, b1;
#pragma unroll
      for (int j = 0; j < 8; j += 2) {
        const int* q = cp0 + 2 * j * Ndim;
        uint32_t a0 = (uint32_t)__builtin_nontemporal_load(q);
        uint32_t a1 = (uint32_t)__builtin_nontemporal_load(q + Ndim);
        uint32_t a2 = (uint32_t)__builtin_nontemporal_load(q + 2 * Ndim);
        uint32_t a3 = (uint32_t)__builtin_nontemporal_load(q + 3 * Ndim);
        dequad(a0, a1, a2, a3, hs0, b0.h[j], b0.h[j + 1]);
        uint32_t c0 = (uint32_t)__builtin_nontemporal_load(q + 16);
        uint32_t c1 = (uint32_t)__builtin_nontemporal_load(q + 16 + Ndim);
        uint32_t c2 = (uint32_t)__builtin_nontemporal_load(q + 16 + 2 * Ndim);
        uint32_t c3 = (uint32_t)__builtin_nontemporal_load(q + 16 + 3 * Ndim);
        dequad(c0, c1, c2, c3, hs1, b1.h[j], b1.h[j + 1]);
      }

      acc0 = __builtin_amdgcn_wmma_f32_16x16x32_f16(false, af.v, false, b0.v,
                                                    (short)0, acc0, false, false);
      acc1 = __builtin_amdgcn_wmma_f32_16x16x32_f16(false, af.v, false, b1.v,
                                                    (short)0, acc1, false, false);
    }
  }

  // ---- deterministic split-K reduction via LDS
  // D layout: VGPR r, lanes0-15 -> row r, col lane; lanes16-31 -> row r+8, col lane-16
#pragma unroll
  for (int r = 0; r < 8; ++r) {
    const int row = r + 8 * hx;
    red[wid][row * 32 + lo16]      = acc0[r];
    red[wid][row * 32 + 16 + lo16] = acc1[r];
  }
  __syncthreads();

  for (int idx = threadIdx.x; idx < 512; idx += 256) {
    float sum = 0.0f;
#pragma unroll
    for (int w = 0; w < 8; ++w) sum += red[w][idx];
    const int row = idx >> 5, col = idx & 31;
    __builtin_nontemporal_store(sum, out + row * Ndim + n0 + col);
  }
}

extern "C" void kernel_launch(void* const* d_in, const int* in_sizes, int n_in,
                              void* d_out, int out_size, void* d_ws, size_t ws_size,
                              hipStream_t stream) {
  const float* A     = (const float*)d_in[0];
  const int*   codes = (const int*)d_in[1];
  const float* s     = (const float*)d_in[2];
  float*       out   = (float*)d_out;
  dim3 grid(12288 / 32);   // 384 WGs x 8 waves = 3072 waves
  marlin_fp4_wmma<<<grid, 256, 0, stream>>>(A, codes, s, out);
}
